// TopologyGNN_50422916055274
// MI455X (gfx1250) — compile-verified, matrix operand order checked by reading, not run on hardware
//
#include <hip/hip_runtime.h>
#include <hip/hip_bf16.h>

// ---------------- types ----------------
typedef __attribute__((ext_vector_type(16))) __bf16 v16bf;
typedef __attribute__((ext_vector_type(8)))  __bf16 v8bf;
typedef __attribute__((ext_vector_type(8)))  float  v8f;

union VU { v16bf v; v8bf h[2]; };

__device__ __forceinline__ __bf16 f2bf(float f) {
  union { float f; unsigned u; } in; in.f = f;
  unsigned u = in.u;
  unsigned r = u + 0x7FFFu + ((u >> 16) & 1u);   // round to nearest even
  union { unsigned short s; __bf16 b; } out;
  out.s = (unsigned short)(r >> 16);
  return out.b;
}

// monotonic uint key for IEEE float (for atomicMax-based segment max)
__device__ __forceinline__ unsigned fkey(float f) {
  unsigned u = __float_as_uint(f);
  return (u & 0x80000000u) ? ~u : (u | 0x80000000u);
}
__device__ __forceinline__ float kinv(unsigned k) {
  unsigned u = (k & 0x80000000u) ? (k & 0x7FFFFFFFu) : ~k;
  return __uint_as_float(u);
}

// ---------------- conversion kernels ----------------
// out[n*K+k] = (bf16) in[k*N+n]   (store W transposed, contiguous in K)
__global__ void k_transpose_bf16(const float* __restrict__ in, __bf16* __restrict__ out,
                                 int K, int N) {
  int idx = blockIdx.x * blockDim.x + threadIdx.x;
  if (idx >= K * N) return;
  int n = idx / K, k = idx % K;
  out[idx] = f2bf(in[(size_t)k * N + n]);
}

__global__ void k_f32_to_bf16(const float* __restrict__ in, __bf16* __restrict__ out,
                              long long total) {
  long long i = (long long)blockIdx.x * blockDim.x + threadIdx.x;
  if (i >= total) return;
  out[i] = f2bf(in[i]);
}

// ---------------- WMMA GEMM: C[M,N] = A[M,K](bf16) * Bt[N,K](bf16)^T ----------------
// Register-tiled: one wave computes a 16(M) x 64(N) strip = 4 accumulators sharing
// one A fragment per K-step (2.5 b128 loads per WMMA instead of 4).
// K multiple of 32; M multiple of 16; N multiple of 64 (nt4 = N/64).
__global__ __launch_bounds__(256)
void k_gemm_bf16_wmma(const __bf16* __restrict__ A, int lda,
                      const __bf16* __restrict__ Bt, int ldb,
                      float* __restrict__ C, int ldc,
                      int mt, int nt4, int K) {
  int wid = (blockIdx.x * blockDim.x + threadIdx.x) >> 5;
  if (wid >= mt * nt4) return;
  int tn = wid % nt4, tm = wid / nt4;
  int lane = threadIdx.x & 31;
  int lm = lane & 15, half = lane >> 4;

  const __bf16* arow = A + (size_t)(tm * 16 + lm) * lda;
  const __bf16* brow = Bt + (size_t)(tn * 64 + lm) * ldb;

  v8f acc[4] = {};
  for (int k0 = 0; k0 < K; k0 += 32) {
    VU a;
    a.h[0] = *(const v8bf*)(arow + k0 +      half * 8);
    a.h[1] = *(const v8bf*)(arow + k0 + 16 + half * 8);
#pragma unroll
    for (int t = 0; t < 4; ++t) {
      VU b;
      const __bf16* bp = brow + (size_t)t * 16 * ldb + k0;
      b.h[0] = *(const v8bf*)(bp +      half * 8);
      b.h[1] = *(const v8bf*)(bp + 16 + half * 8);
      acc[t] = __builtin_amdgcn_wmma_f32_16x16x32_bf16(false, a.v, false, b.v,
                                                       (short)0, acc[t], false, false);
    }
  }
  float* cc = C + (size_t)(tm * 16 + half * 8) * ldc + tn * 64 + lm;
#pragma unroll
  for (int t = 0; t < 4; ++t)
#pragma unroll
    for (int r = 0; r < 8; ++r) cc[(size_t)r * ldc + t * 16] = acc[t][r];
}

// ---------------- attention score projections ----------------
// es[n,h] = sum_c xh[n,h,c]*a_src[h,c] ; ed likewise
__global__ void k_escore(const float* __restrict__ xh, const float* __restrict__ a_src,
                         const float* __restrict__ a_dst, float* __restrict__ es,
                         float* __restrict__ ed, int N, int H, int C) {
  int idx = blockIdx.x * blockDim.x + threadIdx.x;
  if (idx >= N * H) return;
  int n = idx / H, h = idx % H;
  const float* xp = xh + (size_t)n * H * C + (size_t)h * C;
  const float* as = a_src + h * C;
  const float* ad = a_dst + h * C;
  float s0 = 0.f, s1 = 0.f;
  for (int c = 0; c < C; ++c) { float v = xp[c]; s0 += v * as[c]; s1 += v * ad[c]; }
  es[idx] = s0; ed[idx] = s1;
}

// ---------------- edge pass 1: segment max of leaky_relu scores ----------------
__global__ void k_edge_max(const int* __restrict__ ei, const float* __restrict__ es,
                           const float* __restrict__ ed, unsigned* __restrict__ mkey,
                           int E, int N, int H) {
  int e = blockIdx.x * blockDim.x + threadIdx.x;
  int ET = E + N;
  if (e >= ET) return;
  int s = e < E ? ei[e]     : e - E;
  int d = e < E ? ei[E + e] : e - E;
  for (int h = 0; h < H; ++h) {
    float a = es[(size_t)s * H + h] + ed[(size_t)d * H + h];
    a = a > 0.f ? a : 0.2f * a;
    atomicMax(&mkey[(size_t)d * H + h], fkey(a));
  }
}

// ---------------- edge pass 2: p = exp(alpha - max), denom += p ----------------
__global__ void k_edge_exp(const int* __restrict__ ei, const float* __restrict__ es,
                           const float* __restrict__ ed, const unsigned* __restrict__ mkey,
                           float* __restrict__ denom, float* __restrict__ pbuf,
                           int E, int N, int H) {
  int e = blockIdx.x * blockDim.x + threadIdx.x;
  int ET = E + N;
  if (e >= ET) return;
  int s = e < E ? ei[e]     : e - E;
  int d = e < E ? ei[E + e] : e - E;
  for (int h = 0; h < H; ++h) {
    float a = es[(size_t)s * H + h] + ed[(size_t)d * H + h];
    a = a > 0.f ? a : 0.2f * a;
    float m = kinv(mkey[(size_t)d * H + h]);
    float p = expf(a - m);
    pbuf[(size_t)e * H + h] = p;
    atomicAdd(&denom[(size_t)d * H + h], p);
  }
}

// ---------------- edge pass 3: acc[dst] += xh[src] * coef  (wave per edge) ----------------
// float4 (b128) gathers; heads align to float4 boundaries (C in {64,32}).
__global__ __launch_bounds__(256)
void k_edge_agg(const int* __restrict__ ei, const float* __restrict__ xh,
                const float* __restrict__ pbuf, const float* __restrict__ denom,
                float* __restrict__ acc, int E, int N, int H, int cshift) {
  int wid  = (blockIdx.x * blockDim.x + threadIdx.x) >> 5;
  int lane = threadIdx.x & 31;
  int ET = E + N;
  if (wid >= ET) return;
  int s = wid < E ? ei[wid]     : wid - E;
  int d = wid < E ? ei[E + wid] : wid - E;
  float cval = 0.f;
  if (lane < H) cval = pbuf[(size_t)wid * H + lane] / denom[(size_t)d * H + lane];
  int HC  = H << cshift;
  int HC4 = HC >> 2;
  int hs4 = cshift - 2;                       // head = float4_index >> hs4
  const float4* xs4 = (const float4*)(xh  + (size_t)s * HC);
  float*        ad  =                acc + (size_t)d * HC;
  for (int i = lane; i < HC4; i += 32) {
    float c = __shfl(cval, i >> hs4, 32);
    float4 v = xs4[i];
    atomicAdd(&ad[i * 4 + 0], v.x * c);
    atomicAdd(&ad[i * 4 + 1], v.y * c);
    atomicAdd(&ad[i * 4 + 2], v.z * c);
    atomicAdd(&ad[i * 4 + 3], v.w * c);
  }
}

// ---------------- epilogues ----------------
__global__ void k_elu_bias_bf(float* __restrict__ acc, const float* __restrict__ b,
                              __bf16* __restrict__ obf, int HC, long long total) {
  long long i = (long long)blockIdx.x * blockDim.x + threadIdx.x;
  if (i >= total) return;
  float v = acc[i] + b[i % HC];
  v = v > 0.f ? v : (expf(v) - 1.f);     // ELU(alpha=1)
  acc[i] = v;
  obf[i] = f2bf(v);
}

__global__ void k_bias_bf(float* __restrict__ acc, const float* __restrict__ b,
                          __bf16* __restrict__ obf, int HC, long long total) {
  long long i = (long long)blockIdx.x * blockDim.x + threadIdx.x;
  if (i >= total) return;
  float v = acc[i] + b[i % HC];
  acc[i] = v;
  obf[i] = f2bf(v);
}

// ---------------- fused edge MLP: relu(Ptop[s]+Pbot[d]+bm1) -> relu(@Wm2+bm2) -> @Wm3+bm3 -> sigmoid
__global__ __launch_bounds__(256)
void k_edge_mlp(const float* __restrict__ Ptop, const float* __restrict__ Pbot,
                const int* __restrict__ ei, const float* __restrict__ bm1,
                const float* __restrict__ Wm2, const float* __restrict__ bm2,
                const float* __restrict__ Wm3, const float* __restrict__ bm3,
                float* __restrict__ out, int E) {
  int wid  = (blockIdx.x * blockDim.x + threadIdx.x) >> 5;
  int lane = threadIdx.x & 31;
  if (wid >= E) return;
  int s = ei[wid], d = ei[E + wid];
  float h1a = Ptop[(size_t)s * 64 + lane]      + Pbot[(size_t)d * 64 + lane]      + bm1[lane];
  float h1b = Ptop[(size_t)s * 64 + lane + 32] + Pbot[(size_t)d * 64 + lane + 32] + bm1[lane + 32];
  h1a = h1a > 0.f ? h1a : 0.f;
  h1b = h1b > 0.f ? h1b : 0.f;
  float acc = bm2[lane];
#pragma unroll
  for (int j = 0; j < 32; ++j) {
    acc += __shfl(h1a, j, 32) * Wm2[j * 32 + lane];
    acc += __shfl(h1b, j, 32) * Wm2[(j + 32) * 32 + lane];
  }
  float h2 = acc > 0.f ? acc : 0.f;
  float part = h2 * Wm3[lane];
#pragma unroll
  for (int off = 16; off > 0; off >>= 1) part += __shfl_down(part, off, 32);
  if (lane == 0) out[wid] = 1.f / (1.f + expf(-(part + bm3[0])));
}

// ---------------- host launch ----------------
extern "C" void kernel_launch(void* const* d_in, const int* in_sizes, int n_in,
                              void* d_out, int out_size, void* d_ws, size_t ws_size,
                              hipStream_t stream) {
  const int N = 50000, E = 400000, IN = 128;
  const int H1 = 8, C1 = 64, HC1 = 512;
  const int H2 = 4, C2 = 32, HC2 = 128;
  const int ET = E + N;

  const float* x      = (const float*)d_in[0];
  const int*   ei     = (const int*)  d_in[1];
  const float* W1     = (const float*)d_in[2];
  const float* a_src1 = (const float*)d_in[3];
  const float* a_dst1 = (const float*)d_in[4];
  const float* b1     = (const float*)d_in[5];
  const float* W2     = (const float*)d_in[6];
  const float* a_src2 = (const float*)d_in[7];
  const float* a_dst2 = (const float*)d_in[8];
  const float* b2     = (const float*)d_in[9];
  const float* Wm1    = (const float*)d_in[10];
  const float* bm1    = (const float*)d_in[11];
  const float* Wm2    = (const float*)d_in[12];
  const float* bm2    = (const float*)d_in[13];
  const float* Wm3    = (const float*)d_in[14];
  const float* bm3    = (const float*)d_in[15];
  float* out = (float*)d_out;

  // ---- workspace layout (256B aligned regions) ----
  char* ws = (char*)d_ws;
  size_t off = 0;
  auto take = [&](size_t b) { size_t r = off; off += (b + 255) & ~(size_t)255; return r; };

  float* h_acc = (float*)(ws + take((size_t)N * HC1 * 4));   // layer1 accumulator -> h (ELU, in place)
  char*  bigA  =          ws + take((size_t)N * HC1 * 4);    // recycled region
  // phase A: xh1 (f32 [N,512]) occupies all of bigA
  float*  xh1 = (float*)bigA;
  // phase B: hbf | xh2 | z
  __bf16* hbf = (__bf16*)bigA;                                        // [N,512] bf16 = 51.2 MB
  float*  xh2 = (float*)(bigA + (size_t)N * HC1 * 2);                 // [N,128] f32
  float*  z   = (float*)(bigA + (size_t)N * HC1 * 2 + (size_t)N * HC2 * 4);  // [N,128] f32
  // phase C (inside dead hbf span; does not touch z):
  __bf16* zbf  = (__bf16*)bigA;                                       // [N,128] bf16 = 12.8 MB
  float*  Ptop = (float*)(bigA + (size_t)N * HC2 * 2);                // [N,64] f32
  float*  Pbot = (float*)(bigA + (size_t)N * HC2 * 2 + (size_t)N * 64 * 4);

  __bf16* xbf   = (__bf16*)(ws + take((size_t)N * IN * 2));
  __bf16* W1t   = (__bf16*)(ws + take((size_t)512 * 128 * 2));
  __bf16* W2t   = (__bf16*)(ws + take((size_t)128 * 512 * 2));
  __bf16* Wm1t  = (__bf16*)(ws + take((size_t)64 * 256 * 2));
  float*  e_src = (float*) (ws + take((size_t)N * H1 * 4));
  float*  e_dst = (float*) (ws + take((size_t)N * H1 * 4));
  unsigned* mkey= (unsigned*)(ws + take((size_t)N * H1 * 4));
  float*  denom = (float*) (ws + take((size_t)N * H1 * 4));
  float*  pbuf  = (float*) (ws + take((size_t)ET * H1 * 4));

  const int TPB = 256;
  auto cdiv = [](long long a, long long b) { return (int)((a + b - 1) / b); };

  // ---- weight / input conversions ----
  k_transpose_bf16<<<cdiv(128 * 512, TPB), TPB, 0, stream>>>(W1,  W1t,  128, 512);
  k_transpose_bf16<<<cdiv(512 * 128, TPB), TPB, 0, stream>>>(W2,  W2t,  512, 128);
  k_transpose_bf16<<<cdiv(256 * 64,  TPB), TPB, 0, stream>>>(Wm1, Wm1t, 256, 64);
  k_f32_to_bf16<<<cdiv((long long)N * IN, TPB), TPB, 0, stream>>>(x, xbf, (long long)N * IN);

  // ================= GAT layer 1 =================
  // xh1 = x @ W1   (M=50000, N=512, K=128) -> 3125 x 8 strip-tiles
  k_gemm_bf16_wmma<<<cdiv((long long)3125 * 8, 8), TPB, 0, stream>>>(
      xbf, 128, W1t, 128, xh1, 512, 3125, 8, 128);
  k_escore<<<cdiv(N * H1, TPB), TPB, 0, stream>>>(xh1, a_src1, a_dst1, e_src, e_dst, N, H1, C1);

  hipMemsetAsync(mkey,  0, (size_t)N * H1 * 4, stream);
  hipMemsetAsync(denom, 0, (size_t)N * H1 * 4, stream);
  hipMemsetAsync(h_acc, 0, (size_t)N * HC1 * 4, stream);

  k_edge_max<<<cdiv(ET, TPB), TPB, 0, stream>>>(ei, e_src, e_dst, mkey, E, N, H1);
  k_edge_exp<<<cdiv(ET, TPB), TPB, 0, stream>>>(ei, e_src, e_dst, mkey, denom, pbuf, E, N, H1);
  k_edge_agg<<<cdiv(ET, 8), TPB, 0, stream>>>(ei, xh1, pbuf, denom, h_acc, E, N, H1, 6);

  // h = elu(acc + b1); hbf = bf16(h)
  k_elu_bias_bf<<<cdiv((long long)N * HC1, TPB), TPB, 0, stream>>>(
      h_acc, b1, hbf, HC1, (long long)N * HC1);

  // ================= GAT layer 2 =================
  // xh2 = h @ W2   (M=50000, N=128, K=512) -> 3125 x 2 strip-tiles
  k_gemm_bf16_wmma<<<cdiv((long long)3125 * 2, 8), TPB, 0, stream>>>(
      hbf, 512, W2t, 512, xh2, 128, 3125, 2, 512);
  k_escore<<<cdiv(N * H2, TPB), TPB, 0, stream>>>(xh2, a_src2, a_dst2, e_src, e_dst, N, H2, C2);

  hipMemsetAsync(mkey,  0, (size_t)N * H2 * 4, stream);
  hipMemsetAsync(denom, 0, (size_t)N * H2 * 4, stream);
  hipMemsetAsync(z,     0, (size_t)N * HC2 * 4, stream);

  k_edge_max<<<cdiv(ET, TPB), TPB, 0, stream>>>(ei, e_src, e_dst, mkey, E, N, H2);
  k_edge_exp<<<cdiv(ET, TPB), TPB, 0, stream>>>(ei, e_src, e_dst, mkey, denom, pbuf, E, N, H2);
  k_edge_agg<<<cdiv(ET, 8), TPB, 0, stream>>>(ei, xh2, pbuf, denom, z, E, N, H2, 5);

  // z += b2 ; zbf = bf16(z)
  k_bias_bf<<<cdiv((long long)N * HC2, TPB), TPB, 0, stream>>>(
      z, b2, zbf, HC2, (long long)N * HC2);

  // ================= edge MLP =================
  // Ptop = z @ Wm1[0:128,:], Pbot = z @ Wm1[128:256,:]   (M=50000, N=64, K=128) -> 3125 x 1
  k_gemm_bf16_wmma<<<cdiv((long long)3125 * 1, 8), TPB, 0, stream>>>(
      zbf, 128, Wm1t, 256, Ptop, 64, 3125, 1, 128);
  k_gemm_bf16_wmma<<<cdiv((long long)3125 * 1, 8), TPB, 0, stream>>>(
      zbf, 128, Wm1t + 128, 256, Pbot, 64, 3125, 1, 128);

  k_edge_mlp<<<cdiv(E, 8), TPB, 0, stream>>>(Ptop, Pbot, ei, bm1, Wm2, bm2, Wm3, bm3, out, E);
}
